// LRU_53326313947627
// MI455X (gfx1250) — matrix operand order repaired and checked by reading, not compiled, not run on hardware
//
#include <hip/hip_runtime.h>

#define L_SEQ   8192
#define D_DIM   1024
#define NCHUNK  64
#define TCHUNK  128   // L_SEQ / NCHUNK

typedef _Float16 v16h __attribute__((ext_vector_type(16)));
typedef _Float16 v8h  __attribute__((ext_vector_type(8)));
typedef float    v8f  __attribute__((ext_vector_type(8)));

// ---------------------------------------------------------------------------
// Prep: lambda = exp(-exp(nu) + i*exp(theta))
// ---------------------------------------------------------------------------
__global__ void prep_lambda(const float* __restrict__ theta,
                            const float* __restrict__ nu,
                            float* __restrict__ lamR,
                            float* __restrict__ lamI) {
    int c = blockIdx.x * blockDim.x + threadIdx.x;
    if (c >= D_DIM) return;
    float mag = expf(-expf(nu[c]));
    float ph  = expf(theta[c]);
    lamR[c] = mag * cosf(ph);
    lamI[c] = mag * sinf(ph);
}

// Prep: inputs f32 -> f16 (A operand of the GEMM)
__global__ void cvt_inputs(const float* __restrict__ x,
                           _Float16* __restrict__ y, int n) {
    int i = blockIdx.x * blockDim.x + threadIdx.x;
    if (i < n) y[i] = (_Float16)x[i];
}

// Prep: B_norm halves in f16 with exp(gamma_log) row scaling folded in
__global__ void cvt_B(const float* __restrict__ Bre,
                      const float* __restrict__ Bim,
                      const float* __restrict__ glog,
                      _Float16* __restrict__ BreH,
                      _Float16* __restrict__ BimH) {
    int i = blockIdx.x * blockDim.x + threadIdx.x;
    if (i >= D_DIM * D_DIM) return;
    int k = i >> 10;               // output-channel row of B_norm
    float g = expf(glog[k]);
    BreH[i] = (_Float16)(Bre[i] * g);
    BimH[i] = (_Float16)(Bim[i] * g);
}

// ---------------------------------------------------------------------------
// GEMM: Bu[l,k] = sum_d A[l,d] * Bn[k,d]   (two GEMMs: re and im)
//
// Block tile: 128 rows (8 waves x 16) x 64 cols.  B tiles (64 cols x 32 K,
// re+im) are staged in LDS with GLOBAL_LOAD_ASYNC_TO_LDS_B128, double
// buffered: the async copy for K-step i+1 overlaps WMMA of step i.
//
// WMMA fragment layouts (16-bit, 16x16x32; A/B in this code's contiguous K order):
//   A: lane m<16 holds row m, K koff..koff+15 contiguous, koff = 16*(lane>=16)
//   B: lane n<16 holds col n, K 0..15 contiguous; lane n+16 holds K 16..31
//   C: VGPR r -> row r + 8*(lane>=16), col lane&15
// ---------------------------------------------------------------------------
__global__ __launch_bounds__(256) void gemm_wmma(
    const _Float16* __restrict__ A,
    const _Float16* __restrict__ Bre,
    const _Float16* __restrict__ Bim,
    float* __restrict__ BuRe,
    float* __restrict__ BuIm) {

    __shared__ _Float16 sB[2][2][64][32];   // [buf][mat][row(n)][k]  16 KB

    const int tid  = threadIdx.x;
    const int lane = tid & 31;
    const int wave = tid >> 5;
    const int m0   = blockIdx.y * 128 + wave * 16;
    const int n0   = blockIdx.x * 64;

    // ---- async-copy addressing: this thread owns one 16B chunk of the
    // Re tile and one of the Im tile (256 threads x 2 x 16B = 8KB tile) ----
    const int crow  = tid >> 2;   // 0..63  tile row (column n of Bu)
    const int cpart = tid & 3;    // 0..3   16B slice within the 64B row
    const _Float16* gRe = Bre + (size_t)(n0 + crow) * D_DIM + cpart * 8;
    const _Float16* gIm = Bim + (size_t)(n0 + crow) * D_DIM + cpart * 8;
    // generic LDS pointer: low 32 bits == LDS byte address (aperture layout)
    const unsigned ldsRe = (unsigned)(size_t)&sB[0][0][crow][cpart * 8];
    const unsigned ldsIm = (unsigned)(size_t)&sB[0][1][crow][cpart * 8];
    const unsigned bufStride = (unsigned)sizeof(sB[0]);   // 8192 B

    // ---- WMMA fragment addressing ----
    const int arow = m0 + (lane & 15);
    const int koff = (lane >> 4) * 16;
    const int kgrp = (lane >> 4) * 16;
    const int bcol = lane & 15;
    const _Float16* ap = A + (size_t)arow * D_DIM;

    // Plain local arrays, only ever indexed with unroll-time constants:
    // keeps accumulators pinned in registers (no v_mov churn).
    v8f accR[4] = {};
    v8f accI[4] = {};

    auto issue_copy = [&](int buf, int k) {
        const unsigned dR = ldsRe + (unsigned)buf * bufStride;
        const unsigned dI = ldsIm + (unsigned)buf * bufStride;
        const _Float16* sR = gRe + k;
        const _Float16* sI = gIm + k;
        asm volatile("global_load_async_to_lds_b128 %0, %1, off"
                     :: "v"(dR), "v"(sR) : "memory");
        asm volatile("global_load_async_to_lds_b128 %0, %1, off"
                     :: "v"(dI), "v"(sI) : "memory");
    };

    const int NK = D_DIM / 32;   // 32 K-steps

    // One K-step; `cur` is a compile-time constant at each call site so the
    // LDS double-buffer base folds into immediate ds offsets.
    auto step = [&](int i, int cur) {
        const int k = i * 32;
        if (i + 1 < NK) {
            issue_copy(cur ^ 1, k + 32);              // prefetch next tile
            asm volatile("s_wait_asynccnt 0x2" ::: "memory"); // cur copies done
        } else {
            asm volatile("s_wait_asynccnt 0x0" ::: "memory");
        }
        __syncthreads();          // all waves' copies for buf[cur] complete

        // A fragment (per-wave unique rows, direct from global)
        v8h a0 = *(const v8h*)(ap + k + koff);
        v8h a1 = *(const v8h*)(ap + k + koff + 8);
        v16h a = __builtin_shufflevector(a0, a1,
                 0,1,2,3,4,5,6,7,8,9,10,11,12,13,14,15);

        // Load ALL B fragments into distinct registers first so the ds_loads
        // can be clause-batched behind one wait, then stream the 8 WMMAs
        // back-to-back instead of load->wait->wmma per tile.
        v16h brf[4], bif[4];
#pragma unroll
        for (int nt = 0; nt < 4; ++nt) {
            brf[nt] = *(const v16h*)&sB[cur][0][nt * 16 + bcol][kgrp];
            bif[nt] = *(const v16h*)&sB[cur][1][nt * 16 + bcol][kgrp];
        }
#pragma unroll
        for (int nt = 0; nt < 4; ++nt) {
            accR[nt] = __builtin_amdgcn_wmma_f32_16x16x32_f16(
                false, a, false, brf[nt], (short)0, accR[nt], false, false);
            accI[nt] = __builtin_amdgcn_wmma_f32_16x16x32_f16(
                false, a, false, bif[nt], (short)0, accI[nt], false, false);
        }
        __syncthreads();          // everyone done reading buf[cur]
    };

    issue_copy(0, 0);   // prologue: fill buffer 0

    for (int i = 0; i < NK; i += 2) {   // unroll x2: cur constant per half
        step(i, 0);
        step(i + 1, 1);
    }

    // C/D layout: VGPR r -> row m0 + r + 8*(lane>=16), col = lane&15
    const int rbase = m0 + ((lane >> 4) * 8);
#pragma unroll
    for (int nt = 0; nt < 4; ++nt) {
        const int col = n0 + nt * 16 + bcol;
#pragma unroll
        for (int r = 0; r < 8; ++r) {
            const size_t o = (size_t)(rbase + r) * D_DIM + col;
            BuRe[o] = accR[nt][r];
            BuIm[o] = accI[nt][r];
        }
    }
}

// ---------------------------------------------------------------------------
// Scan pass A: per-chunk local recurrence with h0 = 0; emit chunk-final h.
// ---------------------------------------------------------------------------
__global__ void scan_chunk_end(const float* __restrict__ BuRe,
                               const float* __restrict__ BuIm,
                               const float* __restrict__ lamR,
                               const float* __restrict__ lamI,
                               float* __restrict__ endR,
                               float* __restrict__ endI) {
    const int c = blockIdx.y * blockDim.x + threadIdx.x;
    const int j = blockIdx.x;
    const float lr = lamR[c], li = lamI[c];
    float hr = 0.f, hi = 0.f;
    size_t base = (size_t)j * TCHUNK * D_DIM + c;
    for (int t = 0; t < TCHUNK; ++t) {
        const float br = BuRe[base], bi = BuIm[base];
        const float nr = lr * hr - li * hi + br;
        const float ni = lr * hi + li * hr + bi;
        hr = nr; hi = ni;
        base += D_DIM;
    }
    endR[j * D_DIM + c] = hr;
    endI[j * D_DIM + c] = hi;
}

// ---------------------------------------------------------------------------
// Scan pass B: carry_in[j] = lam^T * carry_in[j-1] + end[j-1] (exclusive).
// One block, one thread per channel; lam^128 by 7 complex squarings.
// ---------------------------------------------------------------------------
__global__ void scan_carries(const float* __restrict__ lamR,
                             const float* __restrict__ lamI,
                             const float* __restrict__ endR,
                             const float* __restrict__ endI,
                             float* __restrict__ carR,
                             float* __restrict__ carI) {
    const int c = threadIdx.x;   // blockDim == D_DIM == 1024
    float pr = lamR[c], pi = lamI[c];
#pragma unroll
    for (int s = 0; s < 7; ++s) {        // lam^(2^7) = lam^TCHUNK
        const float nr = pr * pr - pi * pi;
        const float ni = 2.f * pr * pi;
        pr = nr; pi = ni;
    }
    float hr = 0.f, hi = 0.f;
    for (int j = 0; j < NCHUNK; ++j) {
        carR[j * D_DIM + c] = hr;
        carI[j * D_DIM + c] = hi;
        const float er = endR[j * D_DIM + c];
        const float ei = endI[j * D_DIM + c];
        const float nr = pr * hr - pi * hi + er;
        const float ni = pr * hi + pi * hr + ei;
        hr = nr; hi = ni;
    }
}

// ---------------------------------------------------------------------------
// Scan pass C: re-scan chunk seeded with carry; out = Re(h) + inputs.
// ---------------------------------------------------------------------------
__global__ void scan_final(const float* __restrict__ BuRe,
                           const float* __restrict__ BuIm,
                           const float* __restrict__ lamR,
                           const float* __restrict__ lamI,
                           const float* __restrict__ carR,
                           const float* __restrict__ carI,
                           const float* __restrict__ x,
                           float* __restrict__ out) {
    const int c = blockIdx.y * blockDim.x + threadIdx.x;
    const int j = blockIdx.x;
    const float lr = lamR[c], li = lamI[c];
    float hr = carR[j * D_DIM + c];
    float hi = carI[j * D_DIM + c];
    size_t base = (size_t)j * TCHUNK * D_DIM + c;
    for (int t = 0; t < TCHUNK; ++t) {
        const float br = BuRe[base], bi = BuIm[base];
        const float nr = lr * hr - li * hi + br;
        const float ni = lr * hi + li * hr + bi;
        hr = nr; hi = ni;
        out[base] = hr + x[base];
        base += D_DIM;
    }
}

// ---------------------------------------------------------------------------
extern "C" void kernel_launch(void* const* d_in, const int* in_sizes, int n_in,
                              void* d_out, int out_size, void* d_ws, size_t ws_size,
                              hipStream_t stream) {
    const float* inputs = (const float*)d_in[0];
    const float* theta  = (const float*)d_in[1];
    const float* nu     = (const float*)d_in[2];
    const float* glog   = (const float*)d_in[3];
    const float* Bre    = (const float*)d_in[4];
    const float* Bim    = (const float*)d_in[5];
    float* out = (float*)d_out;

    char* ws = (char*)d_ws;
    size_t off = 0;
    auto carve = [&](size_t bytes) -> void* {
        void* p = ws + off;
        off = (off + bytes + 255) & ~(size_t)255;
        return p;
    };

    const size_t LD = (size_t)L_SEQ * D_DIM;
    _Float16* A16  = (_Float16*)carve(LD * 2);
    _Float16* BreH = (_Float16*)carve((size_t)D_DIM * D_DIM * 2);
    _Float16* BimH = (_Float16*)carve((size_t)D_DIM * D_DIM * 2);
    float* BuRe = (float*)carve(LD * 4);
    float* BuIm = (float*)carve(LD * 4);
    float* lamR = (float*)carve(D_DIM * 4);
    float* lamI = (float*)carve(D_DIM * 4);
    float* endR = (float*)carve((size_t)NCHUNK * D_DIM * 4);
    float* endI = (float*)carve((size_t)NCHUNK * D_DIM * 4);
    float* carR = (float*)carve((size_t)NCHUNK * D_DIM * 4);
    float* carI = (float*)carve((size_t)NCHUNK * D_DIM * 4);

    // Prep
    cvt_inputs<<<(int)((LD + 255) / 256), 256, 0, stream>>>(inputs, A16, (int)LD);
    cvt_B<<<(D_DIM * D_DIM + 255) / 256, 256, 0, stream>>>(Bre, Bim, glog, BreH, BimH);
    prep_lambda<<<(D_DIM + 255) / 256, 256, 0, stream>>>(theta, nu, lamR, lamI);

    // GEMM: Bu = inputs @ B_norm^T (re & im) via WMMA + async LDS staging
    gemm_wmma<<<dim3(D_DIM / 64, L_SEQ / 128), 256, 0, stream>>>(
        A16, BreH, BimH, BuRe, BuIm);

    // Chunked associative scan over time
    scan_chunk_end<<<dim3(NCHUNK, D_DIM / 256), 256, 0, stream>>>(
        BuRe, BuIm, lamR, lamI, endR, endI);
    scan_carries<<<1, D_DIM, 0, stream>>>(lamR, lamI, endR, endI, carR, carI);
    scan_final<<<dim3(NCHUNK, D_DIM / 256), 256, 0, stream>>>(
        BuRe, BuIm, lamR, lamI, carR, carI, inputs, out);
}
